// MoEBottleneckB_77610059038894
// MI455X (gfx1250) — compile-verified, matrix operand order checked by reading, not run on hardware
//
#include <hip/hip_runtime.h>
#include <hip/hip_bf16.h>
#include <stdint.h>

typedef __bf16 bf16;
typedef bf16 v16bf __attribute__((ext_vector_type(16)));
typedef bf16 v8bf  __attribute__((ext_vector_type(8)));
typedef float v8f  __attribute__((ext_vector_type(8)));
typedef unsigned int v4u __attribute__((ext_vector_type(4)));
typedef int v4i __attribute__((ext_vector_type(4)));
typedef int v8i __attribute__((ext_vector_type(8)));

#define EPSV 1e-5f
#define BATCH 32
#define CIN   256
#define HH    112
#define WW    112
#define HWSZ  12544
#define MID   64
#define COUT  256
#define EMB   64
#define NT    128
#define NBLK  98   // 12544 / 128

// ---- workspace layout (bytes, all 256-aligned) ----
#define OFF_ACT1 ((size_t)0)             // bf16 [32][64][112][112]  = 51,380,224 B
#define OFF_ACT2 ((size_t)51380224)      // bf16 same                = 51,380,224 B
#define OFF_W1F  ((size_t)102760448)     // bf16 [64][256]           = 32,768 B
#define OFF_W2F  ((size_t)102793216)     // bf16 [9][64][64]         = 73,728 B
#define OFF_W3F  ((size_t)102866944)     // bf16 [256][64]           = 32,768 B
#define OFF_B1   ((size_t)102899712)     // f32 [64]
#define OFF_B2   ((size_t)102899968)     // f32 [64]
#define OFF_IV2  ((size_t)102900224)     // f32 [64]
#define OFF_B3   ((size_t)102900480)     // f32 [256]
#define OFF_GSC  ((size_t)102901504)     // f32 [32][64]

static __device__ __forceinline__ v16bf load_frag(const bf16* p0, const bf16* p1) {
  // ISA 16-bit A/B striping: each lane holds two contiguous 8-element (16B) K runs.
  v8bf lo = *(const v8bf*)p0;
  v8bf hi = *(const v8bf*)p1;
  return __builtin_shufflevector(lo, hi, 0,1,2,3,4,5,6,7,8,9,10,11,12,13,14,15);
}

static __device__ __forceinline__ v8f wmma_bf16(v16bf a, v16bf b, v8f c) {
  return __builtin_amdgcn_wmma_f32_16x16x32_bf16(false, a, false, b, (short)0, c, false, false);
}

// CDNA5 async global->LDS copy (ASYNCcnt path)
static __device__ __forceinline__ void async_b128(uint32_t ldsoff, const void* g) {
  asm volatile("global_load_async_to_lds_b128 %0, %1, off" :: "v"(ldsoff), "v"(g) : "memory");
}
static __device__ __forceinline__ void wait_async0() {
  asm volatile("s_wait_asynccnt 0" ::: "memory");
}

// CDNA5 Tensor Data Mover: copy a 2-D bf16 tile (here: one 16384-elem row) to LDS.
// D# packing per CDNA5 ISA.
static __device__ __forceinline__ void tdm_copy_row_bf16(uint32_t lds_off, const void* gsrc,
                                                         uint32_t nelem) {
  uint64_t ga = (uint64_t)(uintptr_t)gsrc;
  v4u g0;
  g0[0] = 1u;                                    // count=1, user mode, no gather
  g0[1] = lds_off;                               // lds_addr
  g0[2] = (uint32_t)ga;                          // global_addr[31:0]
  g0[3] = (uint32_t)((ga >> 32) & 0x01FFFFFFu) | (2u << 30);  // addr[56:32], type=2
  v8i g1;
  g1[0] = (int)(1u << 16);                       // workgroup_mask=0, data_size=1 (2B)
  g1[1] = (int)((nelem & 0xFFFFu) << 16);        // tensor_dim0[15:0]
  g1[2] = (int)((nelem >> 16) | (1u << 16));     // tensor_dim0[31:16], tensor_dim1=1
  g1[3] = (int)((nelem & 0xFFFFu) << 16);        // tile_dim0 = nelem (<=65535)
  g1[4] = (int)1;                                // tile_dim1=1, tile_dim2=0
  g1[5] = (int)nelem;                            // tensor_dim0_stride lo32
  g1[6] = 0;
  g1[7] = 0;
  v4i z4 = {0, 0, 0, 0};
#if defined(__clang_major__) && __clang_major__ >= 23
  v8i z8 = {0, 0, 0, 0, 0, 0, 0, 0};
  __builtin_amdgcn_tensor_load_to_lds(g0, g1, z4, z4, z8, 0);
#else
  __builtin_amdgcn_tensor_load_to_lds(g0, g1, z4, z4, 0);
#endif
}

// ---------------- prep: fold BN into bf16 weights ----------------
__global__ __launch_bounds__(256) void prep_kernel(
    const float* w1, const float* g1, const float* b1, const float* m1, const float* v1,
    const float* w2, const float* g2, const float* b2, const float* m2, const float* v2,
    const float* w3, const float* g3, const float* b3, const float* m3, const float* v3,
    bf16* w1f, bf16* w2f, bf16* w3f,
    float* bias1, float* bias2, float* inv2w, float* bias3) {
  int t = threadIdx.x;
  if (t < MID) {
    float i1 = rsqrtf(v1[t] + EPSV) * g1[t];
    bias1[t] = b1[t] - m1[t] * i1;
    float i2 = rsqrtf(v2[t] + EPSV) * g2[t];
    bias2[t] = b2[t] - m2[t] * i2;
    inv2w[t] = i2;
  }
  if (t < COUT) {
    float i3 = rsqrtf(v3[t] + EPSV) * g3[t];
    bias3[t] = b3[t] - m3[t] * i3;
  }
  for (int idx = t; idx < MID * CIN; idx += 256) {
    int m = idx >> 8;
    float i1 = rsqrtf(v1[m] + EPSV) * g1[m];
    w1f[idx] = (bf16)(w1[idx] * i1);
  }
  // w2 [m][c][3][3] -> tap-major, c-contiguous: w2f[tap][m][c]
  for (int idx = t; idx < 9 * MID * MID; idx += 256) {
    int tap = idx / (MID * MID);
    int rem = idx - tap * MID * MID;
    int m = rem >> 6, c = rem & 63;
    w2f[idx] = (bf16)(w2[m * 576 + c * 9 + tap]);
  }
  for (int idx = t; idx < COUT * MID; idx += 256) {
    int o = idx >> 6;
    float i3 = rsqrtf(v3[o] + EPSV) * g3[o];
    w3f[idx] = (bf16)(w3[idx] * i3);
  }
}

// ---------------- gate: [32,64] = relu(emb @ gate_w^T + b) ----------------
__global__ __launch_bounds__(256) void gate_kernel(const float* emb, const float* gw,
                                                   const float* gb, const float* inv2w,
                                                   float* gate_out, float* gscale) {
  int g = blockIdx.x * 256 + threadIdx.x;
  if (g >= BATCH * MID) return;
  int b = g >> 6, m = g & 63;
  const float* e = emb + b * EMB;
  const float* w = gw + m * EMB;
  float acc = gb[m];
#pragma unroll 8
  for (int k = 0; k < EMB; ++k) acc += e[k] * w[k];
  float gt = fmaxf(acc, 0.f);
  gate_out[g] = gt;
  gscale[g] = gt * inv2w[m];  // fold BN2 scale (gate applied before BN2)
}

// ---------------- conv1: 1x1 256->64 + BN + ReLU, bf16 WMMA GEMM ----------------
__global__ __launch_bounds__(256) void conv1_kernel(const float* x, const bf16* w1f,
                                                    const float* bias1, bf16* act1) {
  __shared__ bf16 sA[MID * 64];    // [m][kc]  8 KB, filled via async-to-LDS
  __shared__ bf16 sB[NT * 64];     // [n][kc] 16 KB (transposed for contiguous frags)
  __shared__ float sBias[MID];
  int t = threadIdx.x;
  int b = blockIdx.y;
  int n0 = blockIdx.x * NT;
  if (t < MID) sBias[t] = bias1[t];

  int wv = t >> 5, lane = t & 31;
  int mt = wv & 3, np = wv >> 2;
  int l16 = lane & 15, half = lane >> 4, koff = half * 8;
  int nl = t & 127, kl = (t >> 7) * 32;

  v8f acc[4] = {};
  for (int k0 = 0; k0 < CIN; k0 += 64) {
    __syncthreads();
    {  // A chunk 64x64 bf16: async global->LDS, 32B per thread
      int idx = t * 16;
      int m = idx >> 6, kc = idx & 63;
      uint32_t ldsa = (uint32_t)(uintptr_t)(&sA[idx]);
      const char* g = (const char*)(w1f + m * CIN + k0 + kc);
      async_b128(ldsa, g);
      async_b128(ldsa + 16, g + 16);
    }
    {  // B chunk: gather 32 fp32 (clause), then cvt+pack, then 4x ds_store_b128
      const float* src = x + (size_t)(b * CIN + k0 + kl) * HWSZ + n0 + nl;
      float f[32];
#pragma unroll
      for (int i = 0; i < 32; ++i) f[i] = src[(size_t)i * HWSZ];
#pragma unroll
      for (int v = 0; v < 4; ++v) {
        v8bf pk;
#pragma unroll
        for (int e = 0; e < 8; ++e) pk[e] = (bf16)f[v * 8 + e];
        *(v8bf*)&sB[nl * 64 + kl + v * 8] = pk;
      }
    }
    wait_async0();
    __syncthreads();
#pragma unroll
    for (int kk = 0; kk < 64; kk += 32) {
      const bf16* pa = &sA[(mt * 16 + l16) * 64 + kk + koff];
      v16bf af = load_frag(pa, pa + 16);
      v16bf bfr[4];
#pragma unroll
      for (int j4 = 0; j4 < 4; ++j4) {
        const bf16* pb = &sB[((np * 4 + j4) * 16 + l16) * 64 + kk + koff];
        bfr[j4] = load_frag(pb, pb + 16);
      }
#pragma unroll
      for (int j4 = 0; j4 < 4; ++j4) acc[j4] = wmma_bf16(af, bfr[j4], acc[j4]);
    }
  }
#pragma unroll
  for (int j4 = 0; j4 < 4; ++j4) {
    int n = n0 + (np * 4 + j4) * 16 + l16;
    bf16* dstb = act1 + (size_t)(b * MID) * HWSZ + n;
#pragma unroll
    for (int r = 0; r < 8; ++r) {
      int m = mt * 16 + r + half * 8;  // C/D layout: lanes 0-15 M=r, 16-31 M=r+8
      float v = acc[j4][r] + sBias[m];
      dstb[(size_t)m * HWSZ] = (bf16)fmaxf(v, 0.f);
    }
  }
}

// ------- conv2: 3x3 implicit GEMM + per-(b,m) gate + BN2 + ReLU -------
__global__ __launch_bounds__(256) void conv2_kernel(const bf16* act1, const bf16* w2f,
                                                    const float* gscale, const float* bias2,
                                                    bf16* act2) {
  __shared__ bf16 sAct[180 * 64];  // halo 10x18, channel-contiguous: 22.5 KB
  __shared__ float sScale[MID], sBias[MID];
  int t = threadIdx.x;
  int b = blockIdx.z;
  int h0 = blockIdx.y * 8;
  int w0 = blockIdx.x * 16;
  if (t < MID) { sScale[t] = gscale[b * MID + t]; sBias[t] = bias2[t]; }
  // halo load: 640 (c,y) row segments of 18 contiguous bf16 each
  for (int idx = t; idx < MID * 10; idx += 256) {
    int c = idx / 10, y = idx - (idx / 10) * 10;
    int gh = h0 + y - 1;
    bool hok = (gh >= 0 && gh < HH);
    const bf16* srow = act1 + (size_t)(b * MID + c) * HWSZ + (size_t)gh * WW;
#pragma unroll
    for (int xg = 0; xg < 18; ++xg) {
      int gw = w0 + xg - 1;
      bf16 v = (bf16)0.f;
      if (hok && gw >= 0 && gw < WW) v = srow[gw];
      sAct[(y * 18 + xg) * 64 + c] = v;
    }
  }
  __syncthreads();

  int wv = t >> 5, lane = t & 31;
  int mt = wv & 3, np = wv >> 2;
  int l16 = lane & 15, half = lane >> 4, koff = half * 8;
  v8f acc[4] = {};
#pragma unroll
  for (int tap = 0; tap < 9; ++tap) {
    int kh = tap / 3, kw = tap - kh * 3;
    const bf16* pa = w2f + ((size_t)tap * MID + mt * 16 + l16) * 64 + koff;
    v16bf af0 = load_frag(pa, pa + 16);        // ck = 0
    v16bf af1 = load_frag(pa + 32, pa + 48);   // ck = 32
#pragma unroll
    for (int ck = 0; ck < 2; ++ck) {
      v16bf af = ck ? af1 : af0;
      v16bf bfr[4];
#pragma unroll
      for (int j4 = 0; j4 < 4; ++j4) {
        int dy = np * 4 + j4;
        const bf16* pb = &sAct[((dy + kh) * 18 + l16 + kw) * 64 + ck * 32 + koff];
        bfr[j4] = load_frag(pb, pb + 16);
      }
#pragma unroll
      for (int j4 = 0; j4 < 4; ++j4) acc[j4] = wmma_bf16(af, bfr[j4], acc[j4]);
    }
  }
#pragma unroll
  for (int j4 = 0; j4 < 4; ++j4) {
    int dy = np * 4 + j4;
    size_t nbase = (size_t)(h0 + dy) * WW + w0 + l16;
#pragma unroll
    for (int r = 0; r < 8; ++r) {
      int m = mt * 16 + r + half * 8;
      float v = acc[j4][r] * sScale[m] + sBias[m];
      act2[(size_t)(b * MID + m) * HWSZ + nbase] = (bf16)fmaxf(v, 0.f);
    }
  }
}

// ------- conv3: 1x1 64->256 + BN3 + residual + ReLU (fp32 out) -------
__global__ __launch_bounds__(256) void conv3_kernel(const bf16* act2, const bf16* w3f,
                                                    const float* bias3, const float* x,
                                                    float* out) {
  __shared__ bf16 sA[COUT * 64];  // 32 KB, filled by Tensor Data Mover
  __shared__ bf16 sB[NT * 64];    // 16 KB
  __shared__ float sBias[COUT];
  int t = threadIdx.x;
  int b = blockIdx.y;
  int n0 = blockIdx.x * NT;
  if (t < 32) {  // wave 0 issues one TDM descriptor for the whole 32KB panel
    tdm_copy_row_bf16((uint32_t)(uintptr_t)(&sA[0]), w3f, COUT * 64);
  }
  sBias[t] = bias3[t];
  {  // B tile: two-phase gather (clause) then packed ds_store_b128
    int nl = t & 127, kl = (t >> 7) * 32;
    const bf16* src = act2 + (size_t)(b * MID + kl) * HWSZ + n0 + nl;
    bf16 f[32];
#pragma unroll
    for (int i = 0; i < 32; ++i) f[i] = src[(size_t)i * HWSZ];
#pragma unroll
    for (int v = 0; v < 4; ++v) {
      v8bf pk;
#pragma unroll
      for (int e = 0; e < 8; ++e) pk[e] = f[v * 8 + e];
      *(v8bf*)&sB[nl * 64 + kl + v * 8] = pk;
    }
  }
  __builtin_amdgcn_s_wait_tensorcnt(0);
  __syncthreads();

  int wv = t >> 5, lane = t & 31;
  int l16 = lane & 15, half = lane >> 4, koff = half * 8;
  v8f acc[2][8] = {};
#pragma unroll
  for (int kk = 0; kk < 64; kk += 32) {
    v16bf bfr[8];
#pragma unroll
    for (int j = 0; j < 8; ++j) {
      const bf16* pb = &sB[(j * 16 + l16) * 64 + kk + koff];
      bfr[j] = load_frag(pb, pb + 16);
    }
#pragma unroll
    for (int mi = 0; mi < 2; ++mi) {
      const bf16* pa = &sA[((wv * 2 + mi) * 16 + l16) * 64 + kk + koff];
      v16bf af = load_frag(pa, pa + 16);
#pragma unroll
      for (int j = 0; j < 8; ++j) acc[mi][j] = wmma_bf16(af, bfr[j], acc[mi][j]);
    }
  }
#pragma unroll
  for (int mi = 0; mi < 2; ++mi) {
#pragma unroll
    for (int j = 0; j < 8; ++j) {
      int n = n0 + j * 16 + l16;
#pragma unroll
      for (int r = 0; r < 8; ++r) {
        int m = (wv * 2 + mi) * 16 + r + half * 8;
        size_t idx = (size_t)(b * COUT + m) * HWSZ + n;
        float v = acc[mi][j][r] + sBias[m] + x[idx];
        out[idx] = fmaxf(v, 0.f);
      }
    }
  }
}

extern "C" void kernel_launch(void* const* d_in, const int* in_sizes, int n_in,
                              void* d_out, int out_size, void* d_ws, size_t ws_size,
                              hipStream_t stream) {
  (void)in_sizes; (void)n_in; (void)out_size; (void)ws_size;
  const float* x   = (const float*)d_in[0];
  const float* emb = (const float*)d_in[1];
  const float* w1  = (const float*)d_in[2];
  const float* g1  = (const float*)d_in[3];
  const float* b1  = (const float*)d_in[4];
  const float* m1  = (const float*)d_in[5];
  const float* v1  = (const float*)d_in[6];
  const float* w2  = (const float*)d_in[7];
  const float* g2  = (const float*)d_in[8];
  const float* b2  = (const float*)d_in[9];
  const float* m2  = (const float*)d_in[10];
  const float* v2  = (const float*)d_in[11];
  const float* w3  = (const float*)d_in[12];
  const float* g3  = (const float*)d_in[13];
  const float* b3  = (const float*)d_in[14];
  const float* m3  = (const float*)d_in[15];
  const float* v3  = (const float*)d_in[16];
  const float* gw  = (const float*)d_in[17];
  const float* gb  = (const float*)d_in[18];

  char* ws = (char*)d_ws;
  bf16*  act1  = (bf16*)(ws + OFF_ACT1);
  bf16*  act2  = (bf16*)(ws + OFF_ACT2);
  bf16*  w1f   = (bf16*)(ws + OFF_W1F);
  bf16*  w2f   = (bf16*)(ws + OFF_W2F);
  bf16*  w3f   = (bf16*)(ws + OFF_W3F);
  float* bias1 = (float*)(ws + OFF_B1);
  float* bias2 = (float*)(ws + OFF_B2);
  float* inv2w = (float*)(ws + OFF_IV2);
  float* bias3 = (float*)(ws + OFF_B3);
  float* gscale= (float*)(ws + OFF_GSC);

  float* out = (float*)d_out;
  float* gate_out = out + (size_t)BATCH * COUT * HWSZ;  // tuple: (out, gate)

  prep_kernel<<<1, 256, 0, stream>>>(w1, g1, b1, m1, v1, w2, g2, b2, m2, v2,
                                     w3, g3, b3, m3, v3, w1f, w2f, w3f,
                                     bias1, bias2, inv2w, bias3);
  gate_kernel<<<8, 256, 0, stream>>>(emb, gw, gb, inv2w, gate_out, gscale);
  conv1_kernel<<<dim3(NBLK, BATCH), 256, 0, stream>>>(x, w1f, bias1, act1);
  conv2_kernel<<<dim3(7, 14, BATCH), 256, 0, stream>>>(act1, w2f, gscale, bias2, act2);
  conv3_kernel<<<dim3(NBLK, BATCH), 256, 0, stream>>>(act2, w3f, bias3, x, out);
}
